// DeepInterestNetwork_40810779246677
// MI455X (gfx1250) — compile-verified, matrix operand order checked by reading, not run on hardware
//
#include <hip/hip_runtime.h>
#include <hip/hip_bf16.h>
#include <math.h>

typedef __attribute__((ext_vector_type(16))) _Float16 v16h;
typedef __attribute__((ext_vector_type(8)))  _Float16 v8h;
typedef __attribute__((ext_vector_type(8)))  float    v8f;

#define B_SZ 4096
#define T_SZ 200
#define E_SZ 64
#define AH1 64
#define AH2 32
#define MLP_IN_PAD 288   // 272 padded to 9*32
#define MLP_IN_REAL 272
#define MH1 256
#define MH2 128
#define DENSE_SZ 16

// CDNA5 16-bit A-matrix K pattern: lanes 0-15 hold K {0..7,16..23},
// lanes 16-31 hold K {8..15,24..31} (per CDNA5 ISA 7.12.2).
__device__ __forceinline__ int kpat(int e, int hi) {
  return e + ((e >= 8) ? 8 : 0) + 8 * hi;
}

// Concatenate two contiguous 8-half vectors into a WMMA A operand.
__device__ __forceinline__ v16h cat16(v8h lo, v8h hi) {
  return __builtin_shufflevector(lo, hi, 0, 1, 2, 3, 4, 5, 6, 7,
                                         8, 9, 10, 11, 12, 13, 14, 15);
}

// Load one A-fragment (16 halfs per lane) from a row-major f16 LDS tile.
__device__ __forceinline__ v16h load_afrag(const _Float16* rowp, int hi) {
  v8h lo = *(const v8h*)(rowp + 8 * hi);
  v8h hh = *(const v8h*)(rowp + 16 + 8 * hi);
  return cat16(lo, hh);
}

// ---------------------------------------------------------------------------
// Kernel 1: local activation unit. One wave = TWO 16-row tiles of flattened
// B*T (B-fragments amortized over 2 WMMAs; keys/q staged as f16; feature
// math done as packed-f16 vector ops -> v_pk_sub_f16 / v_pk_mul_f16).
// layer1: (32 x 256) @ (256 x 64)  -> 64 wmma / wave
// layer2: (32 x 64)  @ (64 x 32)   -> 8 wmma / wave
// layer3: relu(h2) @ wo (+bo)      -> LDS reduction, 32 scores / wave
// ---------------------------------------------------------------------------
__global__ __launch_bounds__(128)
void din_attn_kernel(const int* __restrict__ target_item,
                     const int* __restrict__ history_items,
                     const float* __restrict__ item_table,
                     const float* __restrict__ att_w1, const float* __restrict__ att_b1,
                     const float* __restrict__ att_w2, const float* __restrict__ att_b2,
                     const float* __restrict__ att_wo, const float* __restrict__ att_bo,
                     float* __restrict__ scores)
{
  __shared__ __align__(32) _Float16 w1s[32 * 512];      // frag-swizzled (32 KB)
  __shared__ __align__(32) _Float16 w2s[4 * 512];       // (4 KB)
  __shared__ __align__(32) _Float16 keyss[4][32 * 64];  // f16 key rows   (16 KB)
  __shared__ __align__(32) _Float16 qss[4][32 * 64];    // f16 query rows (16 KB)
  __shared__ __align__(32) _Float16 h1s[4][32 * 64];    // layer-1 out    (16 KB)
  __shared__ float scs[4][32 * 16];                     // head partials  (8 KB)

  const int tid  = threadIdx.x;
  const int lane = tid & 31;
  const int wave = tid >> 5;
  const int hi   = lane >> 4;
  const int row  = lane & 15;
  const int col  = lane & 15;

  // stage weights once per block, pre-swizzled into WMMA B-fragment order
  for (int idx = tid; idx < 32 * 512; idx += 128) {
    int e = idx & 15, ln = (idx >> 4) & 31, frag = idx >> 9;
    int kk = frag >> 2, nt = frag & 3;
    int K = 32 * kk + kpat(e, ln >> 4);
    int N = nt * 16 + (ln & 15);
    w1s[idx] = (_Float16)att_w1[K * AH1 + N];
  }
  for (int idx = tid; idx < 4 * 512; idx += 128) {
    int e = idx & 15, ln = (idx >> 4) & 31, frag = idx >> 9;
    int kk = frag >> 1, nt = frag & 1;
    int K = 32 * kk + kpat(e, ln >> 4);
    int N = nt * 16 + (ln & 15);
    w2s[idx] = (_Float16)att_w2[K * AH2 + N];
  }
  __syncthreads();

  const int m0 = (blockIdx.x * 4 + wave) * 32;   // 32 rows per wave

  // gather 32 key rows + matching query rows, convert f32->f16 at stage time
  _Float16* keysW = keyss[wave];
  _Float16* qW    = qss[wave];
  for (int r = 0; r < 32; ++r) {
    int m = m0 + r;                 // wave-uniform -> scalar loads for indices
    int b = m / T_SZ;
    int hidx = history_items[m];
    int qidx = target_item[b];
    float2 kf = *(const float2*)&item_table[(size_t)hidx * 64 + 2 * lane];
    float2 qf = *(const float2*)&item_table[(size_t)qidx * 64 + 2 * lane];
    keysW[r * 64 + 2 * lane]     = (_Float16)kf.x;
    keysW[r * 64 + 2 * lane + 1] = (_Float16)kf.y;
    qW[r * 64 + 2 * lane]        = (_Float16)qf.x;
    qW[r * 64 + 2 * lane + 1]    = (_Float16)qf.y;
  }
  __syncthreads();

  // ---- layer 1: features [k | q | k-q | k*q] (K=256) built on the fly ----
  v8f acc1[2][4] = {};
  #pragma unroll
  for (int kk = 0; kk < 8; ++kk) {
    const int g  = kk >> 1;          // feature group
    const int c0 = (kk & 1) * 32;    // column offset within group
    v16h a[2];
    #pragma unroll
    for (int t = 0; t < 2; ++t) {
      const _Float16* kr = keysW + (t * 16 + row) * 64 + c0;
      const _Float16* qr = qW    + (t * 16 + row) * 64 + c0;
      if (g == 0) {
        a[t] = load_afrag(kr, hi);
      } else if (g == 1) {
        a[t] = load_afrag(qr, hi);
      } else {
        v8h klo = *(const v8h*)(kr + 8 * hi);
        v8h khi = *(const v8h*)(kr + 16 + 8 * hi);
        v8h qlo = *(const v8h*)(qr + 8 * hi);
        v8h qhi = *(const v8h*)(qr + 16 + 8 * hi);
        if (g == 2) a[t] = cat16(klo - qlo, khi - qhi);   // v_pk_sub_f16
        else        a[t] = cat16(klo * qlo, khi * qhi);   // v_pk_mul_f16
      }
    }
    #pragma unroll
    for (int nt = 0; nt < 4; ++nt) {
      v16h bf = *(const v16h*)(w1s + ((kk * 4 + nt) * 32 + lane) * 16);
      acc1[0][nt] = __builtin_amdgcn_wmma_f32_16x16x32_f16(
          false, a[0], false, bf, (short)0, acc1[0][nt], false, false);
      acc1[1][nt] = __builtin_amdgcn_wmma_f32_16x16x32_f16(
          false, a[1], false, bf, (short)0, acc1[1][nt], false, false);
    }
  }

  // bias + relu -> h1 (fp16 32x64) in LDS, row-major for A-frag reload
  _Float16* h1W = h1s[wave];
  #pragma unroll
  for (int t = 0; t < 2; ++t) {
    #pragma unroll
    for (int nt = 0; nt < 4; ++nt) {
      int N = nt * 16 + col;
      float b1v = att_b1[N];
      #pragma unroll
      for (int r = 0; r < 8; ++r) {
        int M = t * 16 + r + 8 * hi;
        h1W[M * 64 + N] = (_Float16)fmaxf(acc1[t][nt][r] + b1v, 0.0f);
      }
    }
  }
  __syncthreads();

  // ---- layer 2 ----
  v8f acc2[2][2] = {};
  #pragma unroll
  for (int kk = 0; kk < 2; ++kk) {
    v16h a[2];
    #pragma unroll
    for (int t = 0; t < 2; ++t)
      a[t] = load_afrag(h1W + (t * 16 + row) * 64 + 32 * kk, hi);
    #pragma unroll
    for (int nt = 0; nt < 2; ++nt) {
      v16h bf = *(const v16h*)(w2s + ((kk * 2 + nt) * 32 + lane) * 16);
      acc2[0][nt] = __builtin_amdgcn_wmma_f32_16x16x32_f16(
          false, a[0], false, bf, (short)0, acc2[0][nt], false, false);
      acc2[1][nt] = __builtin_amdgcn_wmma_f32_16x16x32_f16(
          false, a[1], false, bf, (short)0, acc2[1][nt], false, false);
    }
  }

  // ---- layer 3: scalar head, reduce across N lanes through LDS ----
  float part[2][8] = {};
  #pragma unroll
  for (int nt = 0; nt < 2; ++nt) {
    int N = nt * 16 + col;
    float b2v = att_b2[N];
    float wov = att_wo[N];
    #pragma unroll
    for (int t = 0; t < 2; ++t)
      #pragma unroll
      for (int r = 0; r < 8; ++r)
        part[t][r] += fmaxf(acc2[t][nt][r] + b2v, 0.0f) * wov;
  }
  float* scW = scs[wave];
  #pragma unroll
  for (int t = 0; t < 2; ++t)
    #pragma unroll
    for (int r = 0; r < 8; ++r)
      scW[(t * 16 + r + 8 * hi) * 16 + col] = part[t][r];
  __syncthreads();
  {
    // all 32 lanes: lane hi selects tile, lane&15 selects row within tile
    float s = att_bo[0];
    #pragma unroll
    for (int j = 0; j < 16; ++j) s += scW[(hi * 16 + row) * 16 + j];
    scores[m0 + hi * 16 + row] = s;
  }
}

// ---------------------------------------------------------------------------
// Kernel 2: masked softmax over T, interest = sum_t w_t * keys_t, and
// assembly of the (padded) 288-wide MLP input row.
// ---------------------------------------------------------------------------
__global__ __launch_bounds__(64)
void din_softmax_interest_kernel(const int* __restrict__ target_item,
                                 const int* __restrict__ history_items,
                                 const unsigned char* __restrict__ history_mask,
                                 const int* __restrict__ sparse_features,
                                 const float* __restrict__ dense_features,
                                 const float* __restrict__ item_table,
                                 const float* __restrict__ user_table,
                                 const float* __restrict__ ctx_table,
                                 const float* __restrict__ scores,
                                 float* __restrict__ mlp_in)
{
  __shared__ float sbuf[T_SZ];
  __shared__ int   hidx[T_SZ];
  const int b = blockIdx.x;
  const int e = threadIdx.x;

  for (int t = e; t < T_SZ; t += 64) {
    float s = scores[(size_t)b * T_SZ + t];
    sbuf[t] = history_mask[(size_t)b * T_SZ + t] ? s : -1e9f;
    hidx[t] = history_items[(size_t)b * T_SZ + t];
  }
  __syncthreads();

  float mx = -INFINITY;
  for (int t = 0; t < T_SZ; ++t) mx = fmaxf(mx, sbuf[t]);
  __syncthreads();
  for (int t = e; t < T_SZ; t += 64) sbuf[t] = __expf(sbuf[t] - mx);
  __syncthreads();

  float ssum = 0.0f;
  for (int t = 0; t < T_SZ; ++t) ssum += sbuf[t];
  float acc = 0.0f;
  for (int t = 0; t < T_SZ; ++t)
    acc += sbuf[t] * item_table[(size_t)hidx[t] * 64 + e];  // coalesced over e
  float interest = acc / ssum;

  int uidx = sparse_features[b * 2 + 0];
  int cidx = sparse_features[b * 2 + 1];
  int qidx = target_item[b];
  float* dst = mlp_in + (size_t)b * MLP_IN_PAD;
  dst[e]       = user_table[(size_t)uidx * 64 + e];
  dst[64 + e]  = ctx_table[(size_t)cidx * 64 + e];
  dst[128 + e] = item_table[(size_t)qidx * 64 + e];
  dst[192 + e] = interest;
  if (e < DENSE_SZ) {
    dst[256 + e] = dense_features[b * DENSE_SZ + e];
    dst[272 + e] = 0.0f;   // zero pad to 288
  }
}

// ---------------------------------------------------------------------------
// Kernel 3: final MLP. One wave = 16 rows of B.
// layer1: (16 x 288) @ (288 x 256) -> 144 wmma (w1 K-slice staged per step)
// layer2: (16 x 256) @ (256 x 128) -> 64 wmma  (w2 fully LDS-resident)
// layer3: relu(h2) @ out_w (+out_b)
// ---------------------------------------------------------------------------
__global__ __launch_bounds__(128)
void din_mlp_kernel(const float* __restrict__ mlp_in,
                    const float* __restrict__ mlp_w1, const float* __restrict__ mlp_b1,
                    const float* __restrict__ mlp_w2, const float* __restrict__ mlp_b2,
                    const float* __restrict__ out_w, const float* __restrict__ out_b,
                    float* __restrict__ out)
{
  __shared__ __align__(32) _Float16 w2s[64 * 512];     // 64 KB, 8kk x 8nt frags
  __shared__ __align__(32) _Float16 w1s[16 * 512];     // 16 KB slice (one k-step)
  __shared__ __align__(32) _Float16 h1s[4][16 * 256];  // 32 KB
  __shared__ float scs[4][16 * 16];

  const int tid  = threadIdx.x;
  const int lane = tid & 31;
  const int wave = tid >> 5;
  const int hi   = lane >> 4;
  const int row  = lane & 15;
  const int col  = lane & 15;
  const int m0   = (blockIdx.x * 4 + wave) * 16;

  // stage w2 once, frag-swizzled
  for (int idx = tid; idx < 64 * 512; idx += 128) {
    int e = idx & 15, ln = (idx >> 4) & 31, frag = idx >> 9;
    int kk = frag >> 3, nt = frag & 7;
    int K = 32 * kk + kpat(e, ln >> 4);
    int N = nt * 16 + (ln & 15);
    w2s[idx] = (_Float16)mlp_w2[K * MH2 + N];
  }

  // ---- layer 1 over 9 K-slices of 32 ----
  v8f acc1[16] = {};
  for (int ks = 0; ks < 9; ++ks) {
    __syncthreads();
    for (int idx = tid; idx < 16 * 512; idx += 128) {
      int e = idx & 15, ln = (idx >> 4) & 31, nt = idx >> 9;
      int K = 32 * ks + kpat(e, ln >> 4);
      int N = nt * 16 + (ln & 15);
      w1s[idx] = (K < MLP_IN_REAL) ? (_Float16)mlp_w1[K * MH1 + N] : (_Float16)0.0f;
    }
    __syncthreads();
    const float* arow = mlp_in + (size_t)(m0 + row) * MLP_IN_PAD + 32 * ks;
    v16h a;
    #pragma unroll
    for (int j = 0; j < 16; ++j) a[j] = (_Float16)arow[kpat(j, hi)];
    #pragma unroll
    for (int nt = 0; nt < 16; ++nt) {
      v16h bf = *(const v16h*)(w1s + (nt * 32 + lane) * 16);
      acc1[nt] = __builtin_amdgcn_wmma_f32_16x16x32_f16(
          false, a, false, bf, (short)0, acc1[nt], false, false);
    }
  }

  // bias + relu -> h1 fp16 (16 x 256) in LDS
  _Float16* h1W = h1s[wave];
  #pragma unroll
  for (int nt = 0; nt < 16; ++nt) {
    int N = nt * 16 + col;
    float bv = mlp_b1[N];
    #pragma unroll
    for (int r = 0; r < 8; ++r) {
      int M = r + 8 * hi;
      h1W[M * 256 + N] = (_Float16)fmaxf(acc1[nt][r] + bv, 0.0f);
    }
  }
  __syncthreads();

  // ---- layer 2 ----
  v8f acc2[8] = {};
  #pragma unroll
  for (int kk = 0; kk < 8; ++kk) {
    v16h a = load_afrag(h1W + row * 256 + 32 * kk, hi);
    #pragma unroll
    for (int nt = 0; nt < 8; ++nt) {
      v16h bf = *(const v16h*)(w2s + ((kk * 8 + nt) * 32 + lane) * 16);
      acc2[nt] = __builtin_amdgcn_wmma_f32_16x16x32_f16(
          false, a, false, bf, (short)0, acc2[nt], false, false);
    }
  }

  // ---- layer 3: scalar head ----
  float part[8] = {};
  #pragma unroll
  for (int nt = 0; nt < 8; ++nt) {
    int N = nt * 16 + col;
    float bv = mlp_b2[N];
    float wv = out_w[N];
    #pragma unroll
    for (int r = 0; r < 8; ++r)
      part[r] += fmaxf(acc2[nt][r] + bv, 0.0f) * wv;
  }
  float* scW = scs[wave];
  #pragma unroll
  for (int r = 0; r < 8; ++r) scW[(r + 8 * hi) * 16 + col] = part[r];
  __syncthreads();
  if (lane < 16) {
    float s = out_b[0];
    #pragma unroll
    for (int j = 0; j < 16; ++j) s += scW[lane * 16 + j];
    out[m0 + lane] = s;
  }
}

// ---------------------------------------------------------------------------
extern "C" void kernel_launch(void* const* d_in, const int* in_sizes, int n_in,
                              void* d_out, int out_size, void* d_ws, size_t ws_size,
                              hipStream_t stream) {
  (void)in_sizes; (void)n_in; (void)out_size; (void)ws_size;
  const int*   target_item     = (const int*)d_in[0];
  const int*   history_items   = (const int*)d_in[1];
  const unsigned char* history_mask = (const unsigned char*)d_in[2];
  const int*   sparse_features = (const int*)d_in[3];
  const float* dense_features  = (const float*)d_in[4];
  const float* item_table = (const float*)d_in[5];
  const float* user_table = (const float*)d_in[6];
  const float* ctx_table  = (const float*)d_in[7];
  const float* att_w1 = (const float*)d_in[8];
  const float* att_b1 = (const float*)d_in[9];
  const float* att_w2 = (const float*)d_in[10];
  const float* att_b2 = (const float*)d_in[11];
  const float* att_wo = (const float*)d_in[12];
  const float* att_bo = (const float*)d_in[13];
  const float* mlp_w1 = (const float*)d_in[14];
  const float* mlp_b1 = (const float*)d_in[15];
  const float* mlp_w2 = (const float*)d_in[16];
  const float* mlp_b2 = (const float*)d_in[17];
  const float* out_w  = (const float*)d_in[18];
  const float* out_b  = (const float*)d_in[19];

  float* scores = (float*)d_ws;                                          // B*T f32
  float* mlp_in = (float*)((char*)d_ws + (size_t)B_SZ * T_SZ * sizeof(float)); // B*288 f32

  din_attn_kernel<<<(B_SZ * T_SZ) / (32 * 4), 128, 0, stream>>>(
      target_item, history_items, item_table,
      att_w1, att_b1, att_w2, att_b2, att_wo, att_bo, scores);

  din_softmax_interest_kernel<<<B_SZ, 64, 0, stream>>>(
      target_item, history_items, history_mask, sparse_features, dense_features,
      item_table, user_table, ctx_table, scores, mlp_in);

  din_mlp_kernel<<<B_SZ / (16 * 4), 128, 0, stream>>>(
      mlp_in, mlp_w1, mlp_b1, mlp_w2, mlp_b2, out_w, out_b, (float*)d_out);
}
